// SplitContextCenterConv2D_4148938408710
// MI455X (gfx1250) — matrix operand order repaired and checked by reading, not compiled, not run on hardware
//
#include <hip/hip_runtime.h>
#include <hip/hip_bf16.h>

typedef __attribute__((ext_vector_type(16))) __bf16       v16bf;
typedef __attribute__((ext_vector_type(8)))  float        v8f;
typedef __attribute__((ext_vector_type(8)))  unsigned int v8u;

#define H_   256
#define W_   256
#define B_   8
#define Cin_ 256
#define F_   256
#define Ktot 1152   // 9 taps * 128 channels

// f32 -> bf16 RNE (used only in weight pre-pack, off the critical path)
__device__ __forceinline__ unsigned short f2bf(float f) {
  unsigned int u = __float_as_uint(f);
  u += 0x7FFFu + ((u >> 16) & 1u);
  return (unsigned short)(u >> 16);
}
__device__ __forceinline__ float bf2f(unsigned short s) {
  return __uint_as_float(((unsigned int)s) << 16);
}

// Build combined weight matrix Wt[N=256][K=1152] as bf16 hi/lo split.
// K = tap*128 + c. tap 4 (center) takes kernel[1,1,1,c,n]; others kernel[dy,dx,0,c,n].
__global__ __launch_bounds__(256) void pack_weights(
    const float* __restrict__ kern,
    unsigned short* __restrict__ Whi,
    unsigned short* __restrict__ Wlo) {
  int idx = blockIdx.x * blockDim.x + threadIdx.x;
  if (idx >= F_ * Ktot) return;
  int n = idx / Ktot;
  int k = idx - n * Ktot;
  int t = k >> 7;          // tap 0..8 (== dy*3+dx)
  int c = k & 127;
  int g = (t == 4) ? 1 : 0;
  float w = kern[(((size_t)t * 2 + g) * 128 + c) * 256 + n];
  unsigned short h = f2bf(w);
  Whi[idx] = h;
  Wlo[idx] = f2bf(w - bf2f(h));
}

// Implicit-GEMM conv, bf16x3 fp32 emulation.
// Block = 8 waves; block tile = 64(M pixels) x 256(N).
// wave: mi = wave&3 -> 16-pixel M subtile; nh = wave>>2 -> 128-wide N half.
__global__ __launch_bounds__(256) void conv_main(
    const float* __restrict__ x,
    const unsigned short* __restrict__ Whi,
    const unsigned short* __restrict__ Wlo,
    const float* __restrict__ bias,
    float* __restrict__ out) {
  const int lane  = threadIdx.x & 31;
  const int wave  = threadIdx.x >> 5;
  const int mi    = wave & 3;                  // 0..3
  const int nh    = wave >> 2;                 // 0..1
  const int wBase = blockIdx.x * 64 + mi * 16; // pixel column base for this wave
  const int h     = blockIdx.y;
  const int b     = blockIdx.z;
  const int lrow  = lane & 15;
  const int hi16  = lane >> 4;                 // 0 or 1
  const int kA    = hi16 * 16;                 // A-fragment K offset (contiguous K, matches B)
  const int kB    = hi16 * 16;                 // B-fragment K offset (contiguous K, matches A)

  v8f acc[8];
#pragma unroll
  for (int i = 0; i < 8; ++i) acc[i] = v8f{};

#pragma unroll 1
  for (int t = 0; t < 9; ++t) {
    const int dy = t / 3 - 1;
    const int dx = t % 3 - 1;
    const int sh = h + dy;
    const int sw = wBase + lrow + dx;          // this lane's source pixel column
    const bool valid = ((unsigned)sh < (unsigned)H_) && ((unsigned)sw < (unsigned)W_);
    const int shc = valid ? sh : 0;
    const int swc = valid ? sw : 0;
    const int cb  = (t == 4) ? 128 : 0;        // center tap reads channels 128..255
    const float* src =
        x + ((((size_t)b * H_ + shc) * W_ + swc) * Cin_ + cb);
    const int kbaseT = t * 128;

#pragma unroll
    for (int kc = 0; kc < 4; ++kc) {           // 4 x 32-channel K chunks per tap
      float4 q0, q1, q2, q3;
      if (valid) {
        const float* s0 = src + kc * 32 + kA;
        q0 = *(const float4*)(s0);
        q1 = *(const float4*)(s0 + 4);
        q2 = *(const float4*)(s0 + 8);
        q3 = *(const float4*)(s0 + 12);
      } else {
        q0 = make_float4(0.f, 0.f, 0.f, 0.f);
        q1 = q0; q2 = q0; q3 = q0;
      }
      float vals[16] = {q0.x, q0.y, q0.z, q0.w, q1.x, q1.y, q1.z, q1.w,
                        q2.x, q2.y, q2.z, q2.w, q3.x, q3.y, q3.z, q3.w};

      // Truncation split: hi = top16(v) (exact bf16), lo = v - hi (truncated).
      // Pack two bf16 per dword with byte-select-friendly ops (-> v_perm_b32).
      v8u ahw, alw;
#pragma unroll
      for (int d = 0; d < 8; ++d) {
        const unsigned int u0 = __float_as_uint(vals[2 * d]);
        const unsigned int u1 = __float_as_uint(vals[2 * d + 1]);
        const unsigned int h0 = u0 & 0xFFFF0000u;
        const unsigned int h1 = u1 & 0xFFFF0000u;
        const float l0 = vals[2 * d]     - __uint_as_float(h0);
        const float l1 = vals[2 * d + 1] - __uint_as_float(h1);
        ahw[d] = h1 | (u0 >> 16);
        alw[d] = (__float_as_uint(l1) & 0xFFFF0000u) | (__float_as_uint(l0) >> 16);
      }
      const v16bf ahi = __builtin_bit_cast(v16bf, ahw);
      const v16bf alo = __builtin_bit_cast(v16bf, alw);

      const int kbase = kbaseT + kc * 32;
#pragma unroll
      for (int nt = 0; nt < 8; ++nt) {
        const int n = nh * 128 + nt * 16 + lrow;      // B column = lane&15
        const size_t wo = (size_t)n * Ktot + kbase + kB;
        // 16 contiguous K bf16 per lane = one 32B fragment load (2x b128)
        const v16bf bhi = *(const v16bf*)(Whi + wo);
        const v16bf blo = *(const v16bf*)(Wlo + wo);
        // bf16x3 fp32 emulation: D += Ahi*Bhi + Ahi*Blo + Alo*Bhi
        acc[nt] = __builtin_amdgcn_wmma_f32_16x16x32_bf16(
            false, ahi, false, bhi, (short)0, acc[nt], false, false);
        acc[nt] = __builtin_amdgcn_wmma_f32_16x16x32_bf16(
            false, ahi, false, blo, (short)0, acc[nt], false, false);
        acc[nt] = __builtin_amdgcn_wmma_f32_16x16x32_bf16(
            false, alo, false, bhi, (short)0, acc[nt], false, false);
      }
    }
  }

  // Epilogue: C/D layout — VGPR v: lanes0-15 -> M=v, lanes16-31 -> M=v+8; N = lane&15
  const size_t outRow = ((size_t)b * H_ + h) * W_ + wBase;
#pragma unroll
  for (int nt = 0; nt < 8; ++nt) {
    const int f = nh * 128 + nt * 16 + lrow;
    const float bv = bias[f];
#pragma unroll
    for (int v = 0; v < 8; ++v) {
      const int m = v + hi16 * 8;
      out[(outRow + m) * F_ + f] = acc[nt][v] + bv;
    }
  }
}

extern "C" void kernel_launch(void* const* d_in, const int* in_sizes, int n_in,
                              void* d_out, int out_size, void* d_ws, size_t ws_size,
                              hipStream_t stream) {
  const float* x    = (const float*)d_in[0];
  const float* kern = (const float*)d_in[1];
  const float* bias = (const float*)d_in[2];
  float* out = (float*)d_out;

  // workspace: bf16 hi + lo weight matrices, 256*1152*2B each (~1.13 MB total)
  unsigned short* Whi = (unsigned short*)d_ws;
  unsigned short* Wlo = Whi + (size_t)F_ * Ktot;

  const int total = F_ * Ktot;
  pack_weights<<<(total + 255) / 256, 256, 0, stream>>>(kern, Whi, Wlo);

  dim3 grid(W_ / 64, H_, B_);
  conv_main<<<grid, dim3(256, 1, 1), 0, stream>>>(x, Whi, Wlo, bias, out);
}
